// TemporalTX_90005334655900
// MI455X (gfx1250) — matrix operand order repaired and checked by reading, not compiled, not run on hardware
//
#include <hip/hip_runtime.h>
#include <hip/hip_bf16.h>

// ---------------- problem constants ----------------
#define TT   512   // cur_seq
#define BB   16    // batch
#define DD   128   // d_input
#define HH   8     // heads
#define DHH  128   // d_head
#define PP   512   // prev_seq
#define JJ   1024  // TT + PP
#define IB   128   // i-rows per workgroup in attention kernel
#define JS   32    // j-strip width
#define BAND 160   // position band rows per strip (block-wide union)

#define TBD ((size_t)TT * BB * DD)            // 1,048,576
#define NQ  ((size_t)BB * HH * TT * DHH)      // 8,388,608  (bf16 elems)
#define NK  ((size_t)BB * HH * JJ * DHH)      // 16,777,216 (bf16 elems)
#define NP  ((size_t)HH * JJ * DHH)           // 1,048,576  (bf16 elems)

// scale * log2(e): softmax(x * 1/sqrt(DH)) via exp2
#define SCLOG2 0.12751771f
#define NEGBIG (-1e30f)

typedef __attribute__((ext_vector_type(16))) __bf16 v16bf;
typedef __attribute__((ext_vector_type(8)))  float  v8f;

union ABfrag { v16bf v; unsigned u[8]; };

__device__ __forceinline__ unsigned short f2bf(float x) {
  union { float f; unsigned u; } a; a.f = x;
  unsigned r = a.u + 0x7FFFu + ((a.u >> 16) & 1u);   // round-to-nearest-even
  return (unsigned short)(r >> 16);
}

__device__ __forceinline__ v8f wmma_bf16(ABfrag a, ABfrag b, v8f c) {
  return __builtin_amdgcn_wmma_f32_16x16x32_bf16(
      /*neg_a=*/false, a.v, /*neg_b=*/false, b.v,
      /*c_mod=*/(short)0, c, /*reuse_a=*/false, /*reuse_b=*/false);
}

// ---- CDNA5 async global->LDS DMA (ASYNCcnt path, ISA ch10 / 15.18.3) ----
// Low 32 bits of a generic pointer to __shared__ == LDS byte address.
__device__ __forceinline__ void async_cp_b128(unsigned ldsAddr, const void* gaddr) {
  asm volatile("global_load_async_to_lds_b128 %0, %1, off"
               :: "v"(ldsAddr), "v"(gaddr) : "memory");
}
__device__ __forceinline__ void wait_async0() {
  asm volatile("s_wait_asynccnt 0x0" ::: "memory");
}
__device__ __forceinline__ void wait_async14() {
  asm volatile("s_wait_asynccnt 0xe" ::: "memory");
}

// ---------------- kernel A: q projection (+u / +v), bf16 out ----------------
__global__ void qproj_kernel(const float* __restrict__ x,    // [T][B][D]
                             const float* __restrict__ wq,   // [H*DH][D]
                             const float* __restrict__ uu,   // [H][DH]
                             const float* __restrict__ vv,   // [H][DH]
                             unsigned short* __restrict__ qu,// [B][H][T][DH]
                             unsigned short* __restrict__ qv)
{
  size_t idx = (size_t)blockIdx.x * blockDim.x + threadIdx.x;
  if (idx >= NQ) return;
  int dh = (int)(idx % DHH); size_t r = idx / DHH;
  int h  = (int)(r % HH);    r /= HH;
  int b  = (int)(r % BB);    int i = (int)(r / BB);
  int f = h * DHH + dh;
  const float4* xr = (const float4*)(x  + ((size_t)i * BB + b) * DD);
  const float4* wr = (const float4*)(wq + (size_t)f * DD);
  float acc = 0.f;
#pragma unroll 8
  for (int k = 0; k < DD / 4; k++) {
    float4 a = xr[k], w = wr[k];
    acc += a.x * w.x + a.y * w.y + a.z * w.z + a.w * w.w;
  }
  size_t o = (((size_t)b * HH + h) * TT + i) * DHH + dh;
  qu[o] = f2bf(acc + uu[f]);
  qv[o] = f2bf(acc + vv[f]);
}

// ---------------- kernel B: KV projection, bf16 out ----------------
__global__ void kvproj_kernel(const float* __restrict__ x,    // [T][B][D]
                              const float* __restrict__ mem0, // [P][B][D]
                              const float* __restrict__ wkv,  // [2*H*DH][D]
                              unsigned short* __restrict__ Kb,// [B][H][J][DH]
                              unsigned short* __restrict__ Vb)
{
  size_t idx = (size_t)blockIdx.x * blockDim.x + threadIdx.x;
  size_t total = (size_t)JJ * BB * 2 * HH * DHH;
  if (idx >= total) return;
  int f = (int)(idx % (2 * HH * DHH)); size_t r = idx / (2 * HH * DHH);
  int b = (int)(r % BB); int j = (int)(r / BB);
  const float* row = (j < PP) ? (mem0 + ((size_t)j * BB + b) * DD)
                              : (x    + ((size_t)(j - PP) * BB + b) * DD);
  const float4* xr = (const float4*)row;
  const float4* wr = (const float4*)(wkv + (size_t)f * DD);
  float acc = 0.f;
#pragma unroll 8
  for (int k = 0; k < DD / 4; k++) {
    float4 a = xr[k], w = wr[k];
    acc += a.x * w.x + a.y * w.y + a.z * w.z + a.w * w.w;
  }
  if (f < HH * DHH) {
    int h = f / DHH, dh = f % DHH;
    Kb[(((size_t)b * HH + h) * JJ + j) * DHH + dh] = f2bf(acc);
  } else {
    int f2 = f - HH * DHH; int h = f2 / DHH, dh = f2 % DHH;
    Vb[(((size_t)b * HH + h) * JJ + j) * DHH + dh] = f2bf(acc);
  }
}

// ---------------- kernel C: positional projection, bf16 out ----------------
__global__ void pproj_kernel(const float* __restrict__ wp,    // [H*DH][D]
                             unsigned short* __restrict__ Pb) // [H][J][DH]
{
  size_t idx = (size_t)blockIdx.x * blockDim.x + threadIdx.x;
  if (idx >= NP) return;
  int dh = (int)(idx % DHH); size_t r = idx / DHH;
  int jj = (int)(r % JJ);    int h = (int)(r / JJ);
  int f = h * DHH + dh;
  float pos = (float)(JJ - 1 - jj);
  const float* wr = wp + (size_t)f * DD;
  float acc = 0.f;
#pragma unroll 4
  for (int e = 0; e < 64; e++) {
    float invf = __expf(-(float)e * (9.210340371976184f / 64.0f)); // 10000^(-e/64)
    float sp = pos * invf;
    acc += __sinf(sp) * wr[e] + __cosf(sp) * wr[64 + e];
  }
  Pb[((size_t)h * JJ + jj) * DHH + dh] = f2bf(acc);
}

// ---------------- kernel D: fused rel-pos flash attention (WMMA bf16) ----------------
__global__ __launch_bounds__(256, 1)
void attn_kernel(const unsigned short* __restrict__ qu,  // [B][H][T][DH]
                 const unsigned short* __restrict__ qv,
                 const unsigned short* __restrict__ Kb,  // [B][H][J][DH]
                 const unsigned short* __restrict__ Vb,
                 const unsigned short* __restrict__ Pb,  // [H][J][DH]
                 float* __restrict__ ao)                 // [T][B][H*DH]
{
  // double-buffered async-DMA staging, natural [row][dh] layouts
  __shared__ unsigned short Ks[2][JS][DHH];    // 16 KB  K strip
  __shared__ unsigned short Vs[2][JS][DHH];    // 16 KB  V strip
  __shared__ unsigned short Ps[2][BAND][DHH];  // 80 KB  position band window
  // per-wave scratch: rel-shift band (f32) then exp(P) staging (bf16) - disjoint in time
  __shared__ union WScr {
    float          g[16][48];
    unsigned short p[16][32];
  } Scr[8];                                    // 24 KB   (total 136 KB < 320 KB/WG)

  int wg   = blockIdx.x;
  int iblk = wg & 3;
  int bh   = wg >> 2;
  int b = bh >> 3, h = bh & 7;
  int i0   = iblk * IB;
  int lane = threadIdx.x & 31;
  int w    = threadIdx.x >> 5;     // wave 0..7
  int iW   = i0 + w * 16;
  int half = lane >> 4;
  int ln   = lane & 15;

  const unsigned short* Kg = Kb + ((size_t)b * HH + h) * JJ * DHH;
  const unsigned short* Vg = Vb + ((size_t)b * HH + h) * JJ * DHH;
  const unsigned short* Pg = Pb + (size_t)h * JJ * DHH;

  unsigned ksBase = (unsigned)(uintptr_t)&Ks[0][0][0];
  unsigned vsBase = (unsigned)(uintptr_t)&Vs[0][0][0];
  unsigned psBase = (unsigned)(uintptr_t)&Ps[0][0][0];
  const unsigned stripBytes = JS * DHH * 2;      // 8 KB
  const unsigned bandBytes  = BAND * DHH * 2;    // 40 KB

  // ---- load A-fragments for (q+u) and (q+v): 16x32 bf16, 4 k-chunks over DH ----
  ABfrag aqu[4], aqv[4];
  {
    int row = iW + ln;  // A layout: M = lane%16
    const unsigned* pu = (const unsigned*)(qu + (((size_t)b * HH + h) * TT + row) * DHH);
    const unsigned* pv = (const unsigned*)(qv + (((size_t)b * HH + h) * TT + row) * DHH);
#pragma unroll
    for (int kb = 0; kb < 4; kb++)
#pragma unroll
      for (int v = 0; v < 8; v++) {
        int grp = v >> 2, vvk = v & 3;
        int K = kb * 32 + half * 16 + v * 2;   // contiguous k order: K = half*16 + 2v (+1)
        aqu[kb].u[v] = pu[K >> 1];
        aqv[kb].u[v] = pv[K >> 1];
      }
  }

  v8f O[8];
  float mrow[8], lrow[8];
#pragma unroll
  for (int n = 0; n < 8; n++) {
    O[n] = (v8f){0.f, 0.f, 0.f, 0.f, 0.f, 0.f, 0.f, 0.f};
    mrow[n] = NEGBIG; lrow[n] = 0.f;
  }

  int jmax = PP + i0 + IB - 1;
  int nstrips = jmax / JS + 1;

  // per strip, per wave: 4 async ops for K/V (32B per lane each tensor)
  //                    + 10 async ops for the 160-row position band
  unsigned coff = (unsigned)threadIdx.x * 32u;

  // issue K/V strip j0 into buffer `buf`
  auto issue_kv = [&](int j0, int buf) {
    unsigned lb = (unsigned)buf * stripBytes;
    const char* gk = (const char*)(Kg + (size_t)j0 * DHH) + coff;
    const char* gv = (const char*)(Vg + (size_t)j0 * DHH) + coff;
    async_cp_b128(ksBase + lb + coff,      gk);
    async_cp_b128(ksBase + lb + coff + 16, gk + 16);
    async_cp_b128(vsBase + lb + coff,      gv);
    async_cp_b128(vsBase + lb + coff + 16, gv + 16);
  };
  // issue 160-row band window for strip j0 (window base jj = j0 - i0 + 384)
  auto issue_band = [&](int j0, int buf) {
    int base = j0 - i0 + 384;
    unsigned lb = (unsigned)buf * bandBytes;
#pragma unroll
    for (int k = 0; k < 10; k++) {
      int u = k * 256 + (int)threadIdx.x;     // 0..2559 (16B units)
      int r = u >> 4;                         // band row
      int c = u & 15;                         // 16B column chunk
      int jj = base + r;
      jj = jj < 0 ? 0 : (jj > JJ - 1 ? JJ - 1 : jj);   // clamped rows are masked later
      const char* g = (const char*)(Pg + (size_t)jj * DHH) + c * 16;
      async_cp_b128(psBase + lb + (unsigned)u * 16u, g);
    }
  };

  // prologue: stage strip 0 into buffer 0
  issue_kv(0, 0);
  issue_band(0, 0);

  for (int s = 0; s < nstrips; s++) {
    int buf = s & 1;
    int j0 = s * JS;
    if (s + 1 < nstrips) {
      issue_kv(j0 + JS, buf ^ 1);      // overlap next strip's DMA with compute
      issue_band(j0 + JS, buf ^ 1);
      wait_async14();                  // in-order => current strip's 14 DMAs landed
    } else {
      wait_async0();
    }
    __syncthreads();

    // ---- content scores: (q+u) . K^T  -> two 16x16 f32 tiles ----
    v8f c0 = (v8f){0.f,0.f,0.f,0.f,0.f,0.f,0.f,0.f};
    v8f c1 = (v8f){0.f,0.f,0.f,0.f,0.f,0.f,0.f,0.f};
#pragma unroll
    for (int nt = 0; nt < 2; nt++) {
      v8f acc = (v8f){0.f,0.f,0.f,0.f,0.f,0.f,0.f,0.f};
      const unsigned* kr = (const unsigned*)&Ks[buf][nt * 16 + ln][0]; // B: N = lane%16
#pragma unroll
      for (int kb = 0; kb < 4; kb++) {
        ABfrag bf;
#pragma unroll
        for (int v = 0; v < 8; v++) bf.u[v] = kr[kb * 16 + half * 8 + v]; // K = half*16+2v(+1)
        acc = wmma_bf16(aqu[kb], bf, acc);
      }
      if (nt) c1 = acc; else c0 = acc;
    }

    // ---- position band GEMM from LDS window: (q+v) . p[jj]^T ----
    // wave band base within window: (j0 - iW + 496) - (j0 - i0 + 384) = 112 - 16w
    int rbase = 112 - 16 * w;
#pragma unroll
    for (int nt = 0; nt < 3; nt++) {
      v8f acc = (v8f){0.f,0.f,0.f,0.f,0.f,0.f,0.f,0.f};
      const unsigned* pr = (const unsigned*)&Ps[buf][rbase + nt * 16 + ln][0];
#pragma unroll
      for (int kb = 0; kb < 4; kb++) {
        ABfrag bf;
#pragma unroll
        for (int v = 0; v < 8; v++) bf.u[v] = pr[kb * 16 + half * 8 + v];
        acc = wmma_bf16(aqv[kb], bf, acc);
      }
#pragma unroll
      for (int v = 0; v < 8; v++) Scr[w].g[v + 8 * half][nt * 16 + ln] = acc[v];
    }

    // ---- combine, rel-shift gather, mask, scale (log2 units) ----
    float s0[8], s1[8];
#pragma unroll
    for (int v = 0; v < 8; v++) {
      int M = v + 8 * half;
      int i = iW + M;
      {
        int N = ln;
        float sp = Scr[w].g[M][N - M + 15];
        float sv = (c0[v] + sp) * SCLOG2;
        s0[v] = ((j0 + N) <= PP + i) ? sv : NEGBIG;
      }
      {
        int N = 16 + ln;
        float sp = Scr[w].g[M][N - M + 15];
        float sv = (c1[v] + sp) * SCLOG2;
        s1[v] = ((j0 + N) <= PP + i) ? sv : NEGBIG;
      }
    }

    // ---- online softmax: half-wave reductions (row lives in 16 lanes) ----
    float corr[8];
#pragma unroll
    for (int v = 0; v < 8; v++) {
      float mt = fmaxf(s0[v], s1[v]);
      for (int off = 1; off < 16; off <<= 1) mt = fmaxf(mt, __shfl_xor(mt, off));
      float mn = fmaxf(mrow[v], mt);
      corr[v] = exp2f(mrow[v] - mn);
      mrow[v] = mn;
      float p0 = exp2f(s0[v] - mn);
      float p1 = exp2f(s1[v] - mn);
      float rs = p0 + p1;
      for (int off = 1; off < 16; off <<= 1) rs += __shfl_xor(rs, off);
      lrow[v] = lrow[v] * corr[v] + rs;
      int M = v + 8 * half;
      Scr[w].p[M][ln]      = f2bf(p0);   // band values already consumed above
      Scr[w].p[M][16 + ln] = f2bf(p1);
    }

    // ---- O += P (16x32) . V (32x16 per dh chunk) ----
    ABfrag ap;
    {
      const unsigned* pr = (const unsigned*)&Scr[w].p[ln][0]; // A layout: M = lane%16
#pragma unroll
      for (int v = 0; v < 8; v++) {
        int grp = v >> 2, vvk = v & 3;
        int K = half * 16 + v * 2;
        ap.u[v] = pr[K >> 1];
      }
    }
#pragma unroll
    for (int n = 0; n < 8; n++) {
      ABfrag bv;
      int dhc = n * 16 + ln;                 // B layout: N = lane%16 (dh)
#pragma unroll
      for (int v = 0; v < 8; v++) {
        int kk = half * 16 + 2 * v;          // K(j) = half*16 + 2v (+1)
        unsigned lo = Vs[buf][kk][dhc];
        unsigned hi = Vs[buf][kk + 1][dhc];
        bv.u[v] = lo | (hi << 16);
      }
      v8f cacc;
#pragma unroll
      for (int v = 0; v < 8; v++) cacc[v] = O[n][v] * corr[v];
      O[n] = wmma_bf16(ap, bv, cacc);
    }
    __syncthreads();
  }

  // ---- epilogue: normalize and store f32 ----
#pragma unroll
  for (int n = 0; n < 8; n++)
#pragma unroll
    for (int v = 0; v < 8; v++) {
      int M = v + 8 * half;
      int i = iW + M;
      int dh = n * 16 + ln;
      ao[((size_t)i * BB + b) * (HH * DHH) + h * DHH + dh] = O[n][v] / lrow[v];
    }
}

// ---------------- kernel E: output projection; write layer_out + new_mem[1] ----------------
__global__ void outproj_kernel(const float* __restrict__ ao,   // [T][B][H*DH]
                               const float* __restrict__ wout, // [D][H*DH]
                               float* __restrict__ out)        // d_out
{
  size_t idx = (size_t)blockIdx.x * blockDim.x + threadIdx.x;
  if (idx >= TBD) return;
  int e = (int)(idx % DD); size_t r = idx / DD;
  int b = (int)(r % BB);   int i = (int)(r / BB);
  const float4* xr = (const float4*)(ao   + ((size_t)i * BB + b) * (HH * DHH));
  const float4* wr = (const float4*)(wout + (size_t)e * (HH * DHH));
  float acc = 0.f;
#pragma unroll 8
  for (int k = 0; k < (HH * DHH) / 4; k++) {
    float4 a = xr[k], w = wr[k];
    acc += a.x * w.x + a.y * w.y + a.z * w.z + a.w * w.w;
  }
  out[idx] = acc;              // layer_out
  out[2 * TBD + idx] = acc;    // new_mem[1] = layer_out (cs == ps)
}

// ---------------- kernel F: new_mem[0] = inputs (cs == ps) ----------------
__global__ void memcopy_kernel(const float* __restrict__ x, float* __restrict__ out)
{
  size_t idx = (size_t)blockIdx.x * blockDim.x + threadIdx.x;
  if (idx >= TBD) return;
  out[TBD + idx] = x[idx];
}

// ---------------- launch ----------------
extern "C" void kernel_launch(void* const* d_in, const int* in_sizes, int n_in,
                              void* d_out, int out_size, void* d_ws, size_t ws_size,
                              hipStream_t stream) {
  const float* inputs = (const float*)d_in[0];
  const float* memory = (const float*)d_in[1];   // [2][P][B][D]; mem0 = memory
  const float* w_kv   = (const float*)d_in[2];
  const float* w_q    = (const float*)d_in[3];
  const float* w_p    = (const float*)d_in[4];
  const float* w_out  = (const float*)d_in[5];
  const float* u      = (const float*)d_in[6];
  const float* v      = (const float*)d_in[7];
  float* out = (float*)d_out;

  unsigned short* qu = (unsigned short*)d_ws;
  unsigned short* qv = qu + NQ;
  unsigned short* Kb = qv + NQ;
  unsigned short* Vb = Kb + NK;
  unsigned short* Pb = Vb + NK;
  float*          ao = (float*)(Pb + NP);

  qproj_kernel<<<(unsigned)(NQ / 256), 256, 0, stream>>>(inputs, w_q, u, v, qu, qv);
  kvproj_kernel<<<(unsigned)(((size_t)JJ * BB * 2 * HH * DHH) / 256), 256, 0, stream>>>(
      inputs, memory, w_kv, Kb, Vb);
  pproj_kernel<<<(unsigned)(NP / 256), 256, 0, stream>>>(w_p, Pb);
  attn_kernel<<<BB * HH * (TT / IB), 256, 0, stream>>>(qu, qv, Kb, Vb, Pb, ao);
  outproj_kernel<<<(unsigned)(TBD / 256), 256, 0, stream>>>(ao, w_out, out);
  memcopy_kernel<<<(unsigned)(TBD / 256), 256, 0, stream>>>(inputs, out);
}
